// MultiHeadSelfAttention_85899346390
// MI455X (gfx1250) — matrix operand order repaired and checked by reading, not compiled, not run on hardware
//
#include <hip/hip_runtime.h>
#include <hip/hip_bf16.h>

typedef __bf16 bf16;
typedef __attribute__((ext_vector_type(4)))  __bf16 v4bf;
typedef __attribute__((ext_vector_type(8)))  __bf16 v8bf;
typedef __attribute__((ext_vector_type(16))) __bf16 v16bf;
typedef __attribute__((ext_vector_type(8)))  float  v8f;
typedef unsigned int u32;
typedef __attribute__((ext_vector_type(4))) u32 u32x4;
typedef __attribute__((ext_vector_type(4))) int i32x4;
typedef __attribute__((ext_vector_type(8))) int i32x8;

#define B_  4
#define S_  2048
#define D_  1024
#define H_  16
#define DK  64

#if __has_builtin(__builtin_amdgcn_tensor_load_to_lds)
#define HAS_TDM 1
#else
#define HAS_TDM 0
#endif

// ---------------------------------------------------------------------------
// Fragment loaders for v_wmma_f32_16x16x32_bf16 (wave32).
// A (16x32): lane m=lane&15 is row; half=lane>>4 picks interleaved K chunks
//   (K = half*8+0..7 then 16+half*8+0..7).
// B (32x16): lane n=lane&15 is column; half picks interleaved K chunks (same as A).
//   Every B-source is stored so its columns are contiguous rows in memory.
// ---------------------------------------------------------------------------
__device__ __forceinline__ v16bf load_frag_a(const bf16* base, int ld) {
    int lane = threadIdx.x & 31;
    int m = lane & 15, half = lane >> 4;
    const bf16* r = base + (size_t)m * ld + half * 8;
    v8bf lo = *(const v8bf*)(r);
    v8bf hi = *(const v8bf*)(r + 16);
    return __builtin_shufflevector(lo, hi, 0,1,2,3,4,5,6,7,8,9,10,11,12,13,14,15);
}

__device__ __forceinline__ v16bf load_frag_b(const bf16* base, int ld) {
    int lane = threadIdx.x & 31;
    int n = lane & 15, half = lane >> 4;
    const bf16* r = base + (size_t)n * ld + half * 8;
    v8bf lo = *(const v8bf*)(r);
    v8bf hi = *(const v8bf*)(r + 16);
    return __builtin_shufflevector(lo, hi, 0,1,2,3,4,5,6,7,8,9,10,11,12,13,14,15);
}

__device__ __forceinline__ v8f wmma_bf16(v16bf a, v16bf b, v8f c) {
    return __builtin_amdgcn_wmma_f32_16x16x32_bf16(false, a, false, b, (short)0, c,
                                                   false, false);
}

// ---------------------------------------------------------------------------
// TDM: issue a 2D bf16 tile load (tile_d1 rows x tile_d0 cols) into LDS.
// Descriptor packing per CDNA5 ISA 8.3/8.4: group0 = {flags, lds_addr,
// global_addr lo, global_addr hi | type=2}; group1 holds data_size=2B,
// tensor_dim0/1, tile_dim0/1, tensor_dim0_stride.
// ---------------------------------------------------------------------------
__device__ __forceinline__ void tdm_load_2d(void* lds_dst, const bf16* gsrc,
                                            u32 tensor_d0, u32 tile_d0,
                                            u32 tile_d1, u32 d0_stride) {
#if HAS_TDM
    unsigned long long ga = (unsigned long long)gsrc;
    u32x4 g0;
    g0[0] = 1u;                                   // count=1, user descriptor
    g0[1] = (u32)(unsigned long long)lds_dst;     // LDS byte offset (addr[31:0])
    g0[2] = (u32)ga;                              // global_addr[31:0]
    g0[3] = (u32)((ga >> 32) & 0x1FFFFFFu) | 0x80000000u;  // addr[56:32] | type=2
    i32x8 g1;
    g1[0] = 0x00010000;                           // workgroup_mask=0, data_size=2B
    g1[1] = (int)(tensor_d0 << 16);               // tensor_dim0[15:0]
    g1[2] = (int)((tensor_d0 >> 16) | ((u32)S_ << 16));  // dim0 hi | tensor_dim1 lo
    g1[3] = (int)(((u32)S_ >> 16) | (tile_d0 << 16));    // dim1 hi | tile_dim0
    g1[4] = (int)tile_d1;                         // tile_dim1, tile_dim2=0
    g1[5] = (int)d0_stride;                       // tensor_dim0_stride[31:0]
    g1[6] = 0;                                    // stride hi, dim1_stride lo
    g1[7] = 0;
    i32x4 z4 = {0, 0, 0, 0};
#if __clang_major__ >= 23
    i32x8 z8 = {0, 0, 0, 0, 0, 0, 0, 0};
    __builtin_amdgcn_tensor_load_to_lds(g0, g1, z4, z4, z8, 0);
#else
    __builtin_amdgcn_tensor_load_to_lds(g0, g1, z4, z4, 0);
#endif
#endif
}

__device__ __forceinline__ void wait_tensorcnt0() {
#if __has_builtin(__builtin_amdgcn_s_wait_tensorcnt)
    __builtin_amdgcn_s_wait_tensorcnt((short)0);
#else
    asm volatile("s_wait_tensorcnt 0" ::: "memory");
#endif
}
__device__ __forceinline__ void wait_tensorcnt2() {
#if __has_builtin(__builtin_amdgcn_s_wait_tensorcnt)
    __builtin_amdgcn_s_wait_tensorcnt((short)2);
#else
    asm volatile("s_wait_tensorcnt 2" ::: "memory");
#endif
}

// ---------------------------------------------------------------------------
// f32 -> bf16 conversion
// ---------------------------------------------------------------------------
__global__ void cvt_f32_bf16(const float* __restrict__ src, bf16* __restrict__ dst,
                             int n4) {
    int i = blockIdx.x * blockDim.x + threadIdx.x;
    if (i < n4) {
        float4 f = ((const float4*)src)[i];
        v4bf o;
        o[0] = (bf16)f.x; o[1] = (bf16)f.y; o[2] = (bf16)f.z; o[3] = (bf16)f.w;
        ((v4bf*)dst)[i] = o;
    }
}

// ---------------------------------------------------------------------------
// GEMM: C[M x N] = A[M x K] * W[N x K]^T   (M = B_*S_, K = N = D_)
// Wave tile 64x64: 4 A-frags + 4 B-frags -> 16 wmma per K=32 chunk
// (load:wmma = 0.5). Block = 4 waves side by side in N (64x256 per block).
// MODE 0: bf16 [B,H,S,DK] (Q,K) | MODE 1: bf16 [B,H,DK,S] (V^T) | MODE 2: f32 [M,N]
// ---------------------------------------------------------------------------
template <int MODE>
__global__ __launch_bounds__(128) void gemm_bf16(const bf16* __restrict__ A,
                                                 const bf16* __restrict__ W,
                                                 void* __restrict__ out) {
    const int K = D_;
    int wave = threadIdx.x >> 5;
    int row0 = blockIdx.x * 64;
    int colw = blockIdx.y * 256 + wave * 64;

    v8f c[4][4] = {};

    for (int kc = 0; kc < K; kc += 32) {
        v16bf a[4];
#pragma unroll
        for (int rt = 0; rt < 4; ++rt) {
            const bf16* ab = A + (size_t)(row0 + rt * 16) * K + kc;
            if (kc + 32 < K) __builtin_prefetch(ab + 32, 0, 1);
            a[rt] = load_frag_a(ab, K);
        }
#pragma unroll
        for (int nf = 0; nf < 4; ++nf) {
            v16bf bfr = load_frag_b(W + (size_t)(colw + nf * 16) * K + kc, K);
#pragma unroll
            for (int rt = 0; rt < 4; ++rt)
                c[rt][nf] = wmma_bf16(a[rt], bfr, c[rt][nf]);
        }
    }

    int lane = threadIdx.x & 31;
    int nn = lane & 15, half = lane >> 4;
#pragma unroll
    for (int rt = 0; rt < 4; ++rt)
#pragma unroll
        for (int nf = 0; nf < 4; ++nf)
#pragma unroll
            for (int i = 0; i < 8; ++i) {
                int m = row0 + rt * 16 + i + half * 8;
                int e = colw + nf * 16 + nn;
                float v = c[rt][nf][i];
                if (MODE == 0) {
                    int b = m >> 11, s = m & (S_ - 1);
                    int h = e >> 6,  d = e & 63;
                    ((bf16*)out)[(((size_t)(b * H_ + h) * S_ + s) << 6) + d] = (bf16)v;
                } else if (MODE == 1) {
                    int b = m >> 11, s = m & (S_ - 1);
                    int h = e >> 6,  d = e & 63;
                    ((bf16*)out)[((size_t)((b * H_ + h) * DK + d) * S_) + s] = (bf16)v;
                } else {
                    ((float*)out)[(size_t)m * D_ + e] = v;
                }
            }
}

// ---------------------------------------------------------------------------
// RoPE in-place on [B,H,S,DK] bf16; `scale` folds 1/sqrt(dk) into Q.
// ---------------------------------------------------------------------------
__global__ void rope_kernel(bf16* __restrict__ t, float scale) {
    int idx = blockIdx.x * blockDim.x + threadIdx.x;   // B_*H_*S_*32 threads
    int p = idx & 31;
    int bhs = idx >> 5;
    int s = bhs & (S_ - 1);
    size_t base = (size_t)bhs * DK + 2 * p;
    float inv = __powf(10000.0f, -(float)p / 32.0f);
    float ang = (float)s * inv;
    float sn, cs;
    __sincosf(ang, &sn, &cs);
    float e = (float)t[base];
    float o = (float)t[base + 1];
    t[base]     = (bf16)((e * cs - o * sn) * scale);
    t[base + 1] = (bf16)((e * sn + o * cs) * scale);
}

// ---------------------------------------------------------------------------
// Causal flash attention, TDM-fed.
// Block = 128 threads (4 waves) owns a 64-row query tile of one (b,h).
// All waves share 64-key K/V tiles staged in LDS by the Tensor Data Mover
// (wave 0 issues, double buffered, s_wait_tensorcnt), 16 wmma per key block.
// Q (pre-scaled by 0.125 in RoPE): [B,H,S,DK]; K: [B,H,S,DK]; Vt: [B,H,DK,S].
// ---------------------------------------------------------------------------
__global__ __launch_bounds__(128) void flash_attn(const bf16* __restrict__ Q,
                                                  const bf16* __restrict__ Kk,
                                                  const bf16* __restrict__ Vt,
                                                  bf16* __restrict__ O) {
    __shared__ bf16 ktile[2][64 * DK];   // [key][dk]
    __shared__ bf16 vtile[2][DK * 64];   // [dk][key]
    __shared__ bf16 pstage[4][16 * 64];

    int wave = threadIdx.x >> 5;
    int lane = threadIdx.x & 31;
    int nn = lane & 15, half = lane >> 4;
    int qtg = blockIdx.x;                 // 64-row query block
    int h = blockIdx.y, b = blockIdx.z;

    size_t head = (size_t)(b * H_ + h) * S_ * DK;
    const bf16* Kbase = Kk + head;
    const bf16* Vbase = Vt + (size_t)(b * H_ + h) * DK * S_;
    const bf16* Qp = Q + head + (size_t)(qtg * 64 + wave * 16) * DK;

    v16bf qa0 = load_frag_a(Qp, DK);
    v16bf qa1 = load_frag_a(Qp + 32, DK);

    v8f o[4] = {};
    float mr[8], lr[8];
#pragma unroll
    for (int i = 0; i < 8; ++i) { mr[i] = -3.0e38f; lr[i] = 0.0f; }

    bf16* pl = pstage[wave];
    const int nkb = qtg;                  // key blocks 0..qtg (last one masked)

#if HAS_TDM
    if (wave == 0) {
        tdm_load_2d(ktile[0], Kbase, DK, DK, 64, DK);
        tdm_load_2d(vtile[0], Vbase, S_, 64, DK, S_);
    }
#endif

    for (int kb = 0; kb <= nkb; ++kb) {
        int cur = kb & 1;
#if HAS_TDM
        if (wave == 0) {
            if (kb < nkb) {
                tdm_load_2d(ktile[cur ^ 1], Kbase + (size_t)(kb + 1) * 64 * DK,
                            DK, DK, 64, DK);
                tdm_load_2d(vtile[cur ^ 1], Vbase + (size_t)(kb + 1) * 64,
                            S_, 64, DK, S_);
                wait_tensorcnt2();        // tile kb done, kb+1 in flight
            } else {
                wait_tensorcnt0();
            }
        }
#else
        {   // fallback: cooperative vector copy (128 threads)
            const bf16* gk = Kbase + (size_t)kb * 64 * DK;
            for (int t = threadIdx.x; t < 64 * DK / 8; t += 128)
                ((v8bf*)ktile[cur])[t] = ((const v8bf*)gk)[t];
            for (int r = 0; r < DK; ++r)
                for (int t = threadIdx.x; t < 8; t += 128)
                    ((v8bf*)(vtile[cur] + r * 64))[t] =
                        ((const v8bf*)(Vbase + (size_t)r * S_ + kb * 64))[t];
        }
#endif
        __syncthreads();                  // tile kb visible to all waves

        const bf16* kt = ktile[cur];
        const bf16* vt = vtile[cur];

        // S = Q K^T : 4 score fragments (64 keys), 8 wmma
        v8f sc[4];
#pragma unroll
        for (int t = 0; t < 4; ++t) {
            v8f s = {};
            s = wmma_bf16(qa0, load_frag_b(kt + t * 16 * DK, DK), s);
            s = wmma_bf16(qa1, load_frag_b(kt + t * 16 * DK + 32, DK), s);
            sc[t] = s;
        }

        if (kb == nkb) {                  // diagonal block: causal mask
#pragma unroll
            for (int t = 0; t < 4; ++t)
#pragma unroll
                for (int i = 0; i < 8; ++i) {
                    int qidx = qtg * 64 + wave * 16 + i + half * 8;
                    int kidx = kb * 64 + t * 16 + nn;
                    if (kidx > qidx) sc[t][i] = -3.0e38f;
                }
        }

        // online softmax (rows live across 16-lane half-groups)
        float al[8];
#pragma unroll
        for (int i = 0; i < 8; ++i) {
            float cur_m = fmaxf(fmaxf(sc[0][i], sc[1][i]), fmaxf(sc[2][i], sc[3][i]));
            cur_m = fmaxf(cur_m, __shfl_xor(cur_m, 1));
            cur_m = fmaxf(cur_m, __shfl_xor(cur_m, 2));
            cur_m = fmaxf(cur_m, __shfl_xor(cur_m, 4));
            cur_m = fmaxf(cur_m, __shfl_xor(cur_m, 8));
            float mn = fmaxf(mr[i], cur_m);
            al[i] = __expf(mr[i] - mn);
            mr[i] = mn;
            float rs = 0.0f;
#pragma unroll
            for (int t = 0; t < 4; ++t) {
                float p = __expf(sc[t][i] - mn);
                sc[t][i] = p;
                rs += p;
            }
            rs += __shfl_xor(rs, 1);
            rs += __shfl_xor(rs, 2);
            rs += __shfl_xor(rs, 4);
            rs += __shfl_xor(rs, 8);
            lr[i] = lr[i] * al[i] + rs;
        }
#pragma unroll
        for (int nb = 0; nb < 4; ++nb)
#pragma unroll
            for (int i = 0; i < 8; ++i) o[nb][i] *= al[i];

        // stage P (16x64 bf16) -> A fragments
#pragma unroll
        for (int t = 0; t < 4; ++t)
#pragma unroll
            for (int i = 0; i < 8; ++i)
                pl[(i + half * 8) * 64 + t * 16 + nn] = (bf16)sc[t][i];
        asm volatile("s_wait_dscnt 0" ::: "memory");
        v16bf pa0 = load_frag_a(pl, 64);
        v16bf pa1 = load_frag_a(pl + 32, 64);

        // O += P * V : vtile[dk][key] gives contiguous per-lane key runs
#pragma unroll
        for (int nb = 0; nb < 4; ++nb) {
            o[nb] = wmma_bf16(pa0, load_frag_b(vt + nb * 16 * 64, 64), o[nb]);
            o[nb] = wmma_bf16(pa1, load_frag_b(vt + nb * 16 * 64 + 32, 64), o[nb]);
        }
        __syncthreads();                  // all reads done before buffer reuse
    }

    // normalize and store O to [B,S,D] bf16
#pragma unroll
    for (int nb = 0; nb < 4; ++nb)
#pragma unroll
        for (int i = 0; i < 8; ++i) {
            float vv = o[nb][i] / lr[i];
            int srow = qtg * 64 + wave * 16 + i + half * 8;
            O[((size_t)b * S_ + srow) * D_ + h * DK + nb * 16 + nn] = (bf16)vv;
        }
}

// ---------------------------------------------------------------------------
extern "C" void kernel_launch(void* const* d_in, const int* in_sizes, int n_in,
                              void* d_out, int out_size, void* d_ws, size_t ws_size,
                              hipStream_t stream) {
    const float* x  = (const float*)d_in[0];
    const float* Wq = (const float*)d_in[1];
    const float* Wk = (const float*)d_in[2];
    const float* Wv = (const float*)d_in[3];
    const float* Wo = (const float*)d_in[4];

    const size_t NX = (size_t)B_ * S_ * D_;  // 8M elements
    const size_t NW = (size_t)D_ * D_;       // 1M elements

    bf16* xb  = (bf16*)d_ws;
    bf16* wqb = xb  + NX;
    bf16* wkb = wqb + NW;
    bf16* wvb = wkb + NW;
    bf16* wob = wvb + NW;
    bf16* Qb  = wob + NW;
    bf16* Kb  = Qb  + NX;
    bf16* Vtb = Kb  + NX;
    bf16* Ob  = Vtb + NX;

    // 1) f32 -> bf16 conversions
    int n4x = (int)(NX / 4);
    cvt_f32_bf16<<<(n4x + 255) / 256, 256, 0, stream>>>(x, xb, n4x);
    int n4w = (int)(NW / 4);
    cvt_f32_bf16<<<(n4w + 255) / 256, 256, 0, stream>>>(Wq, wqb, n4w);
    cvt_f32_bf16<<<(n4w + 255) / 256, 256, 0, stream>>>(Wk, wkb, n4w);
    cvt_f32_bf16<<<(n4w + 255) / 256, 256, 0, stream>>>(Wv, wvb, n4w);
    cvt_f32_bf16<<<(n4w + 255) / 256, 256, 0, stream>>>(Wo, wob, n4w);

    // 2) QKV projections (WMMA)
    dim3 gg((B_ * S_) / 64, D_ / 256);
    gemm_bf16<0><<<gg, 128, 0, stream>>>(xb, wqb, Qb);
    gemm_bf16<0><<<gg, 128, 0, stream>>>(xb, wkb, Kb);
    gemm_bf16<1><<<gg, 128, 0, stream>>>(xb, wvb, Vtb);

    // 3) RoPE on Q (scaled by 1/sqrt(dk)) and K
    int nr = B_ * H_ * S_ * 32;
    rope_kernel<<<nr / 256, 256, 0, stream>>>(Qb, 0.125f);
    rope_kernel<<<nr / 256, 256, 0, stream>>>(Kb, 1.0f);

    // 4) causal flash attention (TDM-fed, WMMA)
    flash_attn<<<dim3(S_ / 64, H_, B_), 128, 0, stream>>>(Qb, Kb, Vtb, Ob);

    // 5) output projection (WMMA, f32 epilogue to d_out)
    gemm_bf16<2><<<gg, 128, 0, stream>>>(Ob, wob, d_out);
}
